// GPTBlock_34952443855444
// MI455X (gfx1250) — compile-verified, matrix operand order checked by reading, not run on hardware
//
#include <hip/hip_runtime.h>
#include <hip/hip_bf16.h>
#include <stdint.h>

#define B_   2
#define S_   2048
#define D_   2048
#define NH_  16
#define NKV_ 8
#define HD_  128
#define FF_  8192
#define MTOT (B_*S_)
#define EPS_ 1e-6f

// ---- CDNA5 feature probes ---------------------------------------------------
#if defined(__has_builtin)
#  if __has_builtin(__builtin_amdgcn_global_load_async_to_lds_b128)
#    define HAVE_ASYNC 1
#  endif
#  if __has_builtin(__builtin_amdgcn_tensor_load_to_lds)
#    define HAVE_TDM 1
#  endif
#  if __has_builtin(__builtin_amdgcn_cvt_pk_bf16_f32)
#    define HAVE_CVT_PK 1
#  endif
#endif
#ifndef HAVE_ASYNC
#  define HAVE_ASYNC 0
#endif
#ifndef HAVE_TDM
#  define HAVE_TDM 0
#endif
#ifndef HAVE_CVT_PK
#  define HAVE_CVT_PK 0
#endif

#if HAVE_ASYNC
typedef int v4i __attribute__((ext_vector_type(4)));
typedef __attribute__((address_space(1))) v4i g_v4i;
typedef __attribute__((address_space(3))) v4i l_v4i;
__device__ __forceinline__ void async_b128(const float* g, float* l) {
  __builtin_amdgcn_global_load_async_to_lds_b128(
      (g_v4i*)g, (l_v4i*)l, 0, 0);
}
__device__ __forceinline__ void wait_async_le4() {
#if __has_builtin(__builtin_amdgcn_s_wait_asynccnt)
  __builtin_amdgcn_s_wait_asynccnt(4);
#else
  asm volatile("s_wait_asynccnt 0x4" ::: "memory");
#endif
}
__device__ __forceinline__ void wait_async_le8() {
#if __has_builtin(__builtin_amdgcn_s_wait_asynccnt)
  __builtin_amdgcn_s_wait_asynccnt(8);
#else
  asm volatile("s_wait_asynccnt 0x8" ::: "memory");
#endif
}
__device__ __forceinline__ void wait_async_0() {
#if __has_builtin(__builtin_amdgcn_s_wait_asynccnt)
  __builtin_amdgcn_s_wait_asynccnt(0);
#else
  asm volatile("s_wait_asynccnt 0x0" ::: "memory");
#endif
}
#endif

#if HAVE_TDM
typedef unsigned uint32x4 __attribute__((ext_vector_type(4)));
typedef int      int32x8  __attribute__((ext_vector_type(8)));
typedef int      int32x4  __attribute__((ext_vector_type(4)));
__device__ __forceinline__ unsigned lds_offset(const void* p) {
  return (unsigned)(uintptr_t)(__attribute__((address_space(3))) const void*)p;
}
__device__ __forceinline__ void wait_tensor_le1() {
#if __has_builtin(__builtin_amdgcn_s_wait_tensorcnt)
  __builtin_amdgcn_s_wait_tensorcnt(1);
#else
  asm volatile("s_wait_tensorcnt 0x1" ::: "memory");
#endif
}
__device__ __forceinline__ void wait_tensor_0() {
#if __has_builtin(__builtin_amdgcn_s_wait_tensorcnt)
  __builtin_amdgcn_s_wait_tensorcnt(0);
#else
  asm volatile("s_wait_tensorcnt 0x0" ::: "memory");
#endif
}
// 2D TDM load: tile rows x cols (fp32) from row-major tensor -> contiguous LDS
__device__ __forceinline__ void tdm_load_2d(const float* gaddr, unsigned ldsByte,
                                            int cols /*tile_dim0*/, int rows /*tile_dim1*/,
                                            int rowStride /*elements*/,
                                            int tensorD0, int tensorD1) {
  const unsigned long long ga = (unsigned long long)(uintptr_t)gaddr;
  uint32x4 g0;
  g0[0] = 1u;                                              // count=1, user mode
  g0[1] = ldsByte;                                         // lds_addr
  g0[2] = (unsigned)ga;                                    // global_addr[31:0]
  g0[3] = (unsigned)((ga >> 32) & 0x1FFFFFFu) | 0x80000000u; // addr[56:32], type=2
  int32x8 g1;
  g1[0] = 0x20000;                                         // data_size=4B
  g1[1] = (int)(((unsigned)tensorD0 & 0xFFFFu) << 16);     // tensor_dim0 lo16
  g1[2] = (int)((((unsigned)tensorD0 >> 16) & 0xFFFFu) |
                (((unsigned)tensorD1 & 0xFFFFu) << 16));   // dim0 hi16 | dim1 lo16
  g1[3] = (int)((((unsigned)tensorD1 >> 16) & 0xFFFFu) |
                (((unsigned)cols & 0xFFFFu) << 16));       // dim1 hi16 | tile_dim0
  g1[4] = (int)((unsigned)rows & 0xFFFFu);                 // tile_dim1 (tile_dim2=0)
  g1[5] = (int)(unsigned)rowStride;                        // dim0 stride lo32
  g1[6] = 0;
  g1[7] = 0;
  int32x4 gz4 = {0, 0, 0, 0};
  int32x8 gz8 = {0, 0, 0, 0, 0, 0, 0, 0};
  __builtin_amdgcn_tensor_load_to_lds(g0, g1, gz4, gz4, gz8, 0);
}
#endif

typedef __attribute__((ext_vector_type(16))) __bf16 v16bf;
typedef __attribute__((ext_vector_type(8)))  float  v8f;

union BF16Frag { unsigned u[8]; v16bf v; };

#if HAVE_CVT_PK
typedef __attribute__((ext_vector_type(2))) __bf16 v2bf16;
__device__ __forceinline__ unsigned pack_bf16(float a, float b) {
  union { v2bf16 h; unsigned u; } cv;
  cv.h = __builtin_amdgcn_cvt_pk_bf16_f32(a, b);
  return cv.u;
}
__device__ __forceinline__ unsigned short bf16_1(float a) {
  union { v2bf16 h; unsigned short s[2]; } cv;
  cv.h = __builtin_amdgcn_cvt_pk_bf16_f32(a, 0.f);
  return cv.s[0];
}
#else
// Plain fptrunc: lets the backend select the native bf16 conversion on
// gfx1250 (v_cvt_pk_bf16_f32) instead of a software rounding sequence.
__device__ __forceinline__ unsigned pack_bf16(float a, float b) {
  union { __bf16 h[2]; unsigned u; } cv;
  cv.h[0] = (__bf16)a;
  cv.h[1] = (__bf16)b;
  return cv.u;
}
__device__ __forceinline__ unsigned short bf16_1(float a) {
  union { __bf16 h; unsigned short s; } cv;
  cv.h = (__bf16)a;
  return cv.s;
}
#endif

// ---------------------------------------------------------------------------
// Row-wise RMSNorm over D columns. One block (256 threads) per row.
// ---------------------------------------------------------------------------
__global__ __launch_bounds__(256) void rmsnorm_rows(const float* __restrict__ x,
                                                    const float* __restrict__ w,
                                                    float* __restrict__ out, int D) {
  const int row = blockIdx.x;
  const float* xr = x + (size_t)row * D;
  float ss = 0.f;
  for (int i = threadIdx.x; i < D; i += 256) { float v = xr[i]; ss += v * v; }
  #pragma unroll
  for (int o = 1; o < 32; o <<= 1) ss += __shfl_xor(ss, o, 32);
  __shared__ float red[8];
  if ((threadIdx.x & 31) == 0) red[threadIdx.x >> 5] = ss;
  __syncthreads();
  float tot = 0.f;
  #pragma unroll
  for (int i = 0; i < 8; ++i) tot += red[i];
  const float inv = rsqrtf(tot / (float)D + EPS_);
  float* orow = out + (size_t)row * D;
  for (int i = threadIdx.x; i < D; i += 256) orow[i] = xr[i] * inv * w[i];
}

// ---------------------------------------------------------------------------
// WMMA GEMM: C[M,N] = A[M,K] @ W[K,N]  (fp32 in/out, bf16 WMMA compute)
// 256 threads (8 waves) -> 128x128 C tile; wave -> 32x64; K step 32.
// A tile: per-lane global_load_async_to_lds_b128 (ASYNCcnt, double-buffered).
// W tile: one TDM tensor_load_to_lds per tile (TENSORcnt) when available.
// ---------------------------------------------------------------------------
__global__ __launch_bounds__(256) void gemm_bf16(const float* __restrict__ A,
                                                 const float* __restrict__ W,
                                                 float* __restrict__ C,
                                                 int M, int N, int K) {
  const int bm = blockIdx.y * 128;
  const int bn = blockIdx.x * 128;
  const int tid  = threadIdx.x;
  const int wave = tid >> 5, lane = tid & 31;
  const int lr = lane & 15, lg = lane >> 4;
  const int waveM = (wave >> 1) * 32, waveN = (wave & 1) * 64;

  v8f acc[2][4];
  #pragma unroll
  for (int i = 0; i < 2; ++i)
    #pragma unroll
    for (int j = 0; j < 4; ++j) acc[i][j] = v8f{0.f,0.f,0.f,0.f,0.f,0.f,0.f,0.f};

#if HAVE_ASYNC
  __shared__ float laf[2][128][36];   // A tile, rows padded to 144B
  __shared__ float lwf[2][32][128];   // W tile [k][n], rows 512B

  auto issue_A = [&](int k0, int buf) {
    const int row = tid >> 1;
    const int c0 = (tid & 1) * 16;
    const float* src = A + (size_t)(bm + row) * K + k0 + c0;
    float* dst = &laf[buf][row][c0];
    #pragma unroll
    for (int j = 0; j < 4; ++j) async_b128(src + 4 * j, dst + 4 * j);
  };
#if HAVE_TDM
  auto issue_W = [&](int k0, int buf) {
    if (wave == 0)
      tdm_load_2d(W + (size_t)k0 * N + bn, lds_offset(&lwf[buf][0][0]),
                  /*cols*/128, /*rows*/32, /*rowStride*/N,
                  /*tensorD0*/N, /*tensorD1*/K);
  };
#else
  auto issue_W = [&](int k0, int buf) {
    #pragma unroll
    for (int j = 0; j < 4; ++j) {
      const int c = tid + j * 256;
      const int kk = c >> 5, n4 = (c & 31) * 4;
      async_b128(W + (size_t)(k0 + kk) * N + bn + n4, &lwf[buf][kk][n4]);
    }
  };
#endif

  issue_A(0, 0);
  issue_W(0, 0);
  for (int k0 = 0; k0 < K; k0 += 32) {
    const int buf = (k0 >> 5) & 1;
    const bool more = (k0 + 32) < K;
    if (more) { issue_A(k0 + 32, buf ^ 1); issue_W(k0 + 32, buf ^ 1); }
#if HAVE_TDM
    if (more) wait_async_le4(); else wait_async_0();
    if (wave == 0) { if (more) wait_tensor_le1(); else wait_tensor_0(); }
#else
    if (more) wait_async_le8(); else wait_async_0();
#endif
    __syncthreads();                            // tile `buf` visible to all

    BF16Frag afr[2], bfr[4];
    const int cb = lg * 8;                       // float column base
    #pragma unroll
    for (int i = 0; i < 2; ++i) {
      const int m = waveM + i * 16 + lr;
      #pragma unroll
      for (int v = 0; v < 4; ++v) {
        afr[i].u[v]     = pack_bf16(laf[buf][m][cb + 2*v],      laf[buf][m][cb + 2*v + 1]);
        afr[i].u[4 + v] = pack_bf16(laf[buf][m][cb + 16 + 2*v], laf[buf][m][cb + 16 + 2*v + 1]);
      }
    }
    #pragma unroll
    for (int j = 0; j < 4; ++j) {
      const int n = waveN + j * 16 + lr;
      #pragma unroll
      for (int v = 0; v < 4; ++v) {
        bfr[j].u[v]     = pack_bf16(lwf[buf][cb + 2*v][n],      lwf[buf][cb + 2*v + 1][n]);
        bfr[j].u[4 + v] = pack_bf16(lwf[buf][cb + 16 + 2*v][n], lwf[buf][cb + 16 + 2*v + 1][n]);
      }
    }
    #pragma unroll
    for (int i = 0; i < 2; ++i)
      #pragma unroll
      for (int j = 0; j < 4; ++j)
        acc[i][j] = __builtin_amdgcn_wmma_f32_16x16x32_bf16(
            false, afr[i].v, false, bfr[j].v, (short)0, acc[i][j], false, false);
    __syncthreads();                            // all reads of `buf` done
  }
#else
  __shared__ unsigned la[128][17];
  __shared__ unsigned lw[128][17];
  for (int k0 = 0; k0 < K; k0 += 32) {
    __syncthreads();
    {
      const int row = tid >> 1;
      const int kw0 = (tid & 1) * 8;
      const float* src = A + (size_t)(bm + row) * K + k0 + 2 * kw0;
      if (k0 + 32 < K) __builtin_prefetch(src + 32, 0, 1);
      #pragma unroll
      for (int j = 0; j < 8; ++j)
        la[row][kw0 + j] = pack_bf16(src[2 * j], src[2 * j + 1]);
    }
    {
      #pragma unroll
      for (int j = 0; j < 8; ++j) {
        const int idx = tid + j * 256;
        const int n = idx & 127, kw = idx >> 7;
        const float* src = W + (size_t)(k0 + 2 * kw) * N + bn + n;
        lw[n][kw] = pack_bf16(src[0], src[N]);
      }
    }
    __syncthreads();
    BF16Frag afr[2], bfr[4];
    const int kb = lg * 4;
    #pragma unroll
    for (int i = 0; i < 2; ++i) {
      const int m = waveM + i * 16 + lr;
      #pragma unroll
      for (int v = 0; v < 4; ++v) { afr[i].u[v] = la[m][kb + v]; afr[i].u[4 + v] = la[m][kb + 8 + v]; }
    }
    #pragma unroll
    for (int j = 0; j < 4; ++j) {
      const int n = waveN + j * 16 + lr;
      #pragma unroll
      for (int v = 0; v < 4; ++v) { bfr[j].u[v] = lw[n][kb + v]; bfr[j].u[4 + v] = lw[n][kb + 8 + v]; }
    }
    #pragma unroll
    for (int i = 0; i < 2; ++i)
      #pragma unroll
      for (int j = 0; j < 4; ++j)
        acc[i][j] = __builtin_amdgcn_wmma_f32_16x16x32_bf16(
            false, afr[i].v, false, bfr[j].v, (short)0, acc[i][j], false, false);
  }
#endif

  #pragma unroll
  for (int i = 0; i < 2; ++i)
    #pragma unroll
    for (int j = 0; j < 4; ++j)
      #pragma unroll
      for (int r = 0; r < 8; ++r) {
        const int row = bm + waveM + i * 16 + r + 8 * lg;
        const int col = bn + waveN + j * 16 + lr;
        C[(size_t)row * N + col] = acc[i][j][r];
      }
}

// ---------------------------------------------------------------------------
// Per-head RMSNorm + mask select + RoPE for q/k; mask select for v.
// ---------------------------------------------------------------------------
__global__ __launch_bounds__(32) void qkv_post(
    const float* __restrict__ qt, const float* __restrict__ qv,
    const float* __restrict__ kt, const float* __restrict__ kv,
    const float* __restrict__ vt, const float* __restrict__ vv,
    const float* __restrict__ qnw, const float* __restrict__ mmqnw,
    const float* __restrict__ knw, const float* __restrict__ mmknw,
    const float* __restrict__ cosb, const float* __restrict__ sinb,
    const int* __restrict__ mask,
    float* __restrict__ q, float* __restrict__ k, float* __restrict__ v) {
  const int token = blockIdx.x;
  const int u = blockIdx.y;
  const int lane = threadIdx.x;
  const bool vis = mask[token] != 0;

  if (u >= NH_ + NKV_) {
    const int hv = u - NH_ - NKV_;
    const size_t base = (size_t)token * (NKV_ * HD_) + (size_t)hv * HD_;
    #pragma unroll
    for (int j = 0; j < 4; ++j) {
      const int d = lane + 32 * j;
      v[base + d] = vis ? vv[base + d] : vt[base + d];
    }
    return;
  }
  const bool isq = (u < NH_);
  const int head = isq ? u : (u - NH_);
  const int HN = isq ? NH_ : NKV_;
  const size_t base = (size_t)token * (HN * HD_) + (size_t)head * HD_;
  const float* st = (isq ? qt : kt) + base;
  const float* sv = (isq ? qv : kv) + base;
  const float* nt = isq ? qnw : knw;
  const float* nv = isq ? mmqnw : mmknw;

  float a[4], c[4];
  float sst = 0.f, ssv = 0.f;
  #pragma unroll
  for (int j = 0; j < 4; ++j) {
    const int d = lane + 32 * j;
    a[j] = st[d]; c[j] = sv[d];
    sst += a[j] * a[j]; ssv += c[j] * c[j];
  }
  #pragma unroll
  for (int o = 1; o < 32; o <<= 1) {
    sst += __shfl_xor(sst, o, 32);
    ssv += __shfl_xor(ssv, o, 32);
  }
  const float it = rsqrtf(sst / (float)HD_ + EPS_);
  const float iv = rsqrtf(ssv / (float)HD_ + EPS_);
  float sel[4];
  #pragma unroll
  for (int j = 0; j < 4; ++j) {
    const int d = lane + 32 * j;
    sel[j] = vis ? c[j] * iv * nv[d] : a[j] * it * nt[d];
  }
  float* dst = (isq ? q : k) + base;
  const float* cr = cosb + (size_t)token * HD_;
  const float* sr = sinb + (size_t)token * HD_;
  #pragma unroll
  for (int j = 0; j < 4; ++j) {
    const int d = lane + 32 * j;
    const float rh = (j < 2) ? -sel[j + 2] : sel[j - 2];
    dst[d] = sel[j] * cr[d] + rh * sr[d];
  }
}

// ---------------------------------------------------------------------------
// Flash attention, causal, GQA(rep=2). grid (S/64, B*NH), 128 threads.
// ---------------------------------------------------------------------------
__global__ __launch_bounds__(128) void attn_kernel(const float* __restrict__ q,
                                                   const float* __restrict__ k,
                                                   const float* __restrict__ v,
                                                   float* __restrict__ o) {
#if HAVE_ASYNC
  __shared__ float lkf[64][132];            // K tile fp32 [kpos][hd], padded
  __shared__ float lvf[64][128];            // V tile fp32 [kpos][hd]
#else
  __shared__ unsigned lk[64][65];
  __shared__ unsigned lv[128][33];
#endif
  __shared__ unsigned short lps[4][16][68]; // P per wave: [row][kpos] bf16
  const int qt = blockIdx.x;
  const int bh = blockIdx.y;
  const int b = bh / NH_, hh = bh % NH_;
  const int hk = hh / (NH_ / NKV_);
  const int tid = threadIdx.x;
  const int wave = tid >> 5, lane = tid & 31;
  const int lr = lane & 15, lg = lane >> 4;
  const float* qb = q + ((size_t)b * S_) * (NH_ * HD_) + (size_t)hh * HD_;
  const float* kb = k + ((size_t)b * S_) * (NKV_ * HD_) + (size_t)hk * HD_;
  const float* vb = v + ((size_t)b * S_) * (NKV_ * HD_) + (size_t)hk * HD_;
  const int qrow0 = qt * 64 + wave * 16;
  const float scale = 0.08838834764831845f;  // 1/sqrt(128)

  BF16Frag qf[4];
  {
    const float* qrow = qb + (size_t)(qrow0 + lr) * (NH_ * HD_);
    #pragma unroll
    for (int s = 0; s < 4; ++s) {
      const int kwb = s * 16 + lg * 4;
      #pragma unroll
      for (int w2 = 0; w2 < 4; ++w2) {
        const int kw0 = kwb + w2, kw1 = kwb + 8 + w2;
        qf[s].u[w2]     = pack_bf16(qrow[2 * kw0] * scale, qrow[2 * kw0 + 1] * scale);
        qf[s].u[4 + w2] = pack_bf16(qrow[2 * kw1] * scale, qrow[2 * kw1 + 1] * scale);
      }
    }
  }

  float m_i[8], l_i[8];
  v8f accO[8];
  #pragma unroll
  for (int r = 0; r < 8; ++r) { m_i[r] = -1e30f; l_i[r] = 0.f; }
  #pragma unroll
  for (int t = 0; t < 8; ++t) accO[t] = v8f{0.f,0.f,0.f,0.f,0.f,0.f,0.f,0.f};

  for (int kt = 0; kt <= qt; ++kt) {
    const int kbase = kt * 64;
    __syncthreads();
#if HAVE_ASYNC
    #pragma unroll
    for (int j = 0; j < 16; ++j) {
      const int c = tid + j * 128;
      const int row = c >> 5, c4 = (c & 31) * 4;
      async_b128(kb + (size_t)(kbase + row) * (NKV_ * HD_) + c4, &lkf[row][c4]);
    }
    #pragma unroll
    for (int j = 0; j < 16; ++j) {
      const int c = tid + j * 128;
      const int row = c >> 5, c4 = (c & 31) * 4;
      async_b128(vb + (size_t)(kbase + row) * (NKV_ * HD_) + c4, &lvf[row][c4]);
    }
    wait_async_0();
#else
    #pragma unroll
    for (int j = 0; j < 32; ++j) {
      const int idx = tid + j * 128;
      const int kw = idx & 63, kp = idx >> 6;
      const float* src = kb + (size_t)(kbase + kp) * (NKV_ * HD_) + 2 * kw;
      lk[kp][kw] = pack_bf16(src[0], src[1]);
    }
    #pragma unroll
    for (int j = 0; j < 32; ++j) {
      const int idx = tid + j * 128;
      const int hd = idx & 127, kpw = idx >> 7;
      const float v0 = vb[(size_t)(kbase + 2 * kpw) * (NKV_ * HD_) + hd];
      const float v1 = vb[(size_t)(kbase + 2 * kpw + 1) * (NKV_ * HD_) + hd];
      lv[hd][kpw] = pack_bf16(v0, v1);
    }
#endif
    __syncthreads();

    v8f sacc[4];
    #pragma unroll
    for (int j2 = 0; j2 < 4; ++j2) sacc[j2] = v8f{0.f,0.f,0.f,0.f,0.f,0.f,0.f,0.f};
    #pragma unroll
    for (int s = 0; s < 4; ++s) {
      const int kwb = s * 16 + lg * 4;
      #pragma unroll
      for (int j2 = 0; j2 < 4; ++j2) {
        BF16Frag kf;
        const int n = j2 * 16 + lr;
        #pragma unroll
        for (int w2 = 0; w2 < 4; ++w2) {
#if HAVE_ASYNC
          kf.u[w2]     = pack_bf16(lkf[n][2*(kwb + w2)],     lkf[n][2*(kwb + w2) + 1]);
          kf.u[4 + w2] = pack_bf16(lkf[n][2*(kwb + 8 + w2)], lkf[n][2*(kwb + 8 + w2) + 1]);
#else
          kf.u[w2]     = lk[n][kwb + w2];
          kf.u[4 + w2] = lk[n][kwb + 8 + w2];
#endif
        }
        sacc[j2] = __builtin_amdgcn_wmma_f32_16x16x32_bf16(
            false, qf[s].v, false, kf.v, (short)0, sacc[j2], false, false);
      }
    }
    if (kt == qt) {
      #pragma unroll
      for (int j2 = 0; j2 < 4; ++j2)
        #pragma unroll
        for (int r = 0; r < 8; ++r) {
          const int qr = qrow0 + r + 8 * lg;
          const int kp = kbase + j2 * 16 + lr;
          if (kp > qr) sacc[j2][r] = -1e30f;
        }
    }
    float mnew[8], alpha[8], rs[8];
    #pragma unroll
    for (int r = 0; r < 8; ++r) {
      float mx = m_i[r];
      #pragma unroll
      for (int j2 = 0; j2 < 4; ++j2) mx = fmaxf(mx, sacc[j2][r]);
      #pragma unroll
      for (int off = 1; off < 16; off <<= 1) mx = fmaxf(mx, __shfl_xor(mx, off, 32));
      mnew[r] = mx;
      alpha[r] = __expf(m_i[r] - mx);
      m_i[r] = mx;
      rs[r] = 0.f;
    }
    #pragma unroll
    for (int j2 = 0; j2 < 4; ++j2)
      #pragma unroll
      for (int r = 0; r < 8; ++r) {
        const float p = __expf(sacc[j2][r] - mnew[r]);
        sacc[j2][r] = p;
        rs[r] += p;
      }
    #pragma unroll
    for (int r = 0; r < 8; ++r) {
      float t2 = rs[r];
      #pragma unroll
      for (int off = 1; off < 16; off <<= 1) t2 += __shfl_xor(t2, off, 32);
      l_i[r] = l_i[r] * alpha[r] + t2;
      #pragma unroll
      for (int t = 0; t < 8; ++t) accO[t][r] *= alpha[r];
    }
    #pragma unroll
    for (int j2 = 0; j2 < 4; ++j2)
      #pragma unroll
      for (int r = 0; r < 8; ++r)
        lps[wave][r + 8 * lg][j2 * 16 + lr] = bf16_1(sacc[j2][r]);
    __syncthreads();
    #pragma unroll
    for (int s2 = 0; s2 < 2; ++s2) {
      BF16Frag pf;
      const int kwb = s2 * 16 + lg * 4;
      #pragma unroll
      for (int w2 = 0; w2 < 4; ++w2) {
        const int kw0 = kwb + w2, kw1 = kwb + 8 + w2;
        pf.u[w2]     = (unsigned)lps[wave][lr][2 * kw0] | ((unsigned)lps[wave][lr][2 * kw0 + 1] << 16);
        pf.u[4 + w2] = (unsigned)lps[wave][lr][2 * kw1] | ((unsigned)lps[wave][lr][2 * kw1 + 1] << 16);
      }
      #pragma unroll
      for (int t = 0; t < 8; ++t) {
        BF16Frag vf;
        const int n = t * 16 + lr;
        #pragma unroll
        for (int w2 = 0; w2 < 4; ++w2) {
#if HAVE_ASYNC
          vf.u[w2]     = pack_bf16(lvf[2*(kwb + w2)][n],     lvf[2*(kwb + w2) + 1][n]);
          vf.u[4 + w2] = pack_bf16(lvf[2*(kwb + 8 + w2)][n], lvf[2*(kwb + 8 + w2) + 1][n]);
#else
          vf.u[w2]     = lv[n][kwb + w2];
          vf.u[4 + w2] = lv[n][kwb + 8 + w2];
#endif
        }
        accO[t] = __builtin_amdgcn_wmma_f32_16x16x32_bf16(
            false, pf.v, false, vf.v, (short)0, accO[t], false, false);
      }
    }
  }
  #pragma unroll
  for (int r = 0; r < 8; ++r) {
    const int qr = qrow0 + r + 8 * lg;
    const float invl = 1.f / l_i[r];
    float* orow = o + (((size_t)b * S_ + qr) * NH_ + hh) * HD_;
    #pragma unroll
    for (int t = 0; t < 8; ++t) orow[t * 16 + lr] = accO[t][r] * invl;
  }
}

// ---------------------------------------------------------------------------
// Elementwise helpers
// ---------------------------------------------------------------------------
__global__ __launch_bounds__(256) void select_add(const float* __restrict__ base,
                                                  const float* __restrict__ t,
                                                  const float* __restrict__ v,
                                                  const int* __restrict__ mask,
                                                  float* __restrict__ out, int D) {
  const size_t i = (size_t)blockIdx.x * 256 + threadIdx.x;
  const int token = (int)(i / (size_t)D);
  out[i] = base[i] + (mask[token] ? v[i] : t[i]);
}

__global__ __launch_bounds__(256) void silu_mul(float* __restrict__ gate,
                                                const float* __restrict__ up) {
  const size_t i = (size_t)blockIdx.x * 256 + threadIdx.x;
  const float g = gate[i];
  gate[i] = (g / (1.f + __expf(-g))) * up[i];
}

// ---------------------------------------------------------------------------
static inline void launch_gemm(const float* A, const float* W, float* C,
                               int M, int N, int K, hipStream_t s) {
  dim3 grid(N / 128, M / 128);
  gemm_bf16<<<grid, 256, 0, s>>>(A, W, C, M, N, K);
}

extern "C" void kernel_launch(void* const* d_in, const int* in_sizes, int n_in,
                              void* d_out, int out_size, void* d_ws, size_t ws_size,
                              hipStream_t stream) {
  (void)in_sizes; (void)n_in; (void)out_size; (void)ws_size;
  const float* x      = (const float*)d_in[0];
  const float* cosb   = (const float*)d_in[1];
  const float* sinb   = (const float*)d_in[2];
  const int*   mask   = (const int*)d_in[3];
  const float* anw    = (const float*)d_in[4];
  const float* fnw    = (const float*)d_in[5];
  const float* wq     = (const float*)d_in[6];
  const float* wk     = (const float*)d_in[7];
  const float* wv     = (const float*)d_in[8];
  const float* wo     = (const float*)d_in[9];
  const float* mm_wq  = (const float*)d_in[10];
  const float* mm_wk  = (const float*)d_in[11];
  const float* mm_wv  = (const float*)d_in[12];
  const float* mm_wo  = (const float*)d_in[13];
  const float* gate_w = (const float*)d_in[14];
  const float* up_w   = (const float*)d_in[15];
  const float* down_w = (const float*)d_in[16];
  const float* mm_gate_w = (const float*)d_in[17];
  const float* mm_up_w   = (const float*)d_in[18];
  const float* mm_down_w = (const float*)d_in[19];
  const float* qnw    = (const float*)d_in[20];
  const float* knw    = (const float*)d_in[21];
  const float* mmqnw  = (const float*)d_in[22];
  const float* mmknw  = (const float*)d_in[23];

  float* ws = (float*)d_ws;
  size_t off = 0;
  auto alloc = [&](size_t n) { float* p = ws + off; off += n; return p; };
  const size_t MD = (size_t)MTOT * D_;
  const size_t MKV = (size_t)MTOT * NKV_ * HD_;
  const size_t MFF = (size_t)MTOT * FF_;
  float* h    = alloc(MD);
  float* qtb  = alloc(MD);
  float* qvb  = alloc(MD);
  float* ktb  = alloc(MKV);
  float* kvb  = alloc(MKV);
  float* vtb  = alloc(MKV);
  float* vvb  = alloc(MKV);
  float* qb   = alloc(MD);
  float* kbuf = alloc(MKV);
  float* vbuf = alloc(MKV);
  float* attn = alloc(MD);
  float* ot   = alloc(MD);
  float* ov   = alloc(MD);
  float* h2   = alloc(MD);
  float* g    = alloc(MD);
  float* gt   = alloc(MFF);
  float* gv   = alloc(MFF);
  float* ut   = alloc(MFF);
  float* uv   = alloc(MFF);
  float* fft  = alloc(MD);
  float* ffv  = alloc(MD);

  rmsnorm_rows<<<MTOT, 256, 0, stream>>>(x, anw, h, D_);
  launch_gemm(h, wq,    qtb, MTOT, NH_ * HD_, D_, stream);
  launch_gemm(h, mm_wq, qvb, MTOT, NH_ * HD_, D_, stream);
  launch_gemm(h, wk,    ktb, MTOT, NKV_ * HD_, D_, stream);
  launch_gemm(h, mm_wk, kvb, MTOT, NKV_ * HD_, D_, stream);
  launch_gemm(h, wv,    vtb, MTOT, NKV_ * HD_, D_, stream);
  launch_gemm(h, mm_wv, vvb, MTOT, NKV_ * HD_, D_, stream);
  qkv_post<<<dim3(MTOT, NH_ + 2 * NKV_), 32, 0, stream>>>(
      qtb, qvb, ktb, kvb, vtb, vvb, qnw, mmqnw, knw, mmknw,
      cosb, sinb, mask, qb, kbuf, vbuf);
  attn_kernel<<<dim3(S_ / 64, B_ * NH_), 128, 0, stream>>>(qb, kbuf, vbuf, attn);
  launch_gemm(attn, wo,    ot, MTOT, D_, NH_ * HD_, stream);
  launch_gemm(attn, mm_wo, ov, MTOT, D_, NH_ * HD_, stream);
  select_add<<<(unsigned)(MD / 256), 256, 0, stream>>>(x, ot, ov, mask, h2, D_);
  rmsnorm_rows<<<MTOT, 256, 0, stream>>>(h2, fnw, g, D_);
  launch_gemm(g, gate_w,    gt, MTOT, FF_, D_, stream);
  launch_gemm(g, mm_gate_w, gv, MTOT, FF_, D_, stream);
  launch_gemm(g, up_w,      ut, MTOT, FF_, D_, stream);
  launch_gemm(g, mm_up_w,   uv, MTOT, FF_, D_, stream);
  silu_mul<<<(unsigned)(MFF / 256), 256, 0, stream>>>(gt, ut);
  silu_mul<<<(unsigned)(MFF / 256), 256, 0, stream>>>(gv, uv);
  launch_gemm(gt, down_w,    fft, MTOT, D_, FF_, stream);
  launch_gemm(gv, mm_down_w, ffv, MTOT, D_, FF_, stream);
  select_add<<<(unsigned)(MD / 256), 256, 0, stream>>>(h2, fft, ffv, mask,
                                                       (float*)d_out, D_);
}
